// LLaMAAttention_54992761258542
// MI455X (gfx1250) — matrix operand order repaired and checked by reading, not compiled, not run on hardware
//
#include <hip/hip_runtime.h>
#include <hip/hip_bf16.h>

// ---------------------------------------------------------------------------
// LLaMA attention block for MI455X (gfx1250, wave32, WMMA).
// All GEMM-like math in f16 via v_wmma_f32_16x16x32_f16 (fp32 accum).
// Weights are pre-transposed so every WMMA B-fragment is two b128 loads.
// ---------------------------------------------------------------------------

#define B_   2
#define S_   2048
#define D_   2048
#define H_   16
#define KH_  4
#define HD_  128

typedef __attribute__((ext_vector_type(16))) _Float16 v16h;
typedef __attribute__((ext_vector_type(8)))  _Float16 v8h;
typedef __attribute__((ext_vector_type(8)))  float    v8f;

union AFrag { v16h v; v8h h[2]; };

__device__ __forceinline__ float rowMax16(float x) {
    // xor masks 1..8 reduce within each 16-lane half of the wave32 —
    // exactly the row grouping of the 16x16 C-matrix layout.
    #pragma unroll
    for (int m = 1; m < 16; m <<= 1) x = fmaxf(x, __shfl_xor(x, m, 32));
    return x;
}
__device__ __forceinline__ float rowSum16(float x) {
    #pragma unroll
    for (int m = 1; m < 16; m <<= 1) x += __shfl_xor(x, m, 32);
    return x;
}

// ---------------------------------------------------------------------------
// fp32 -> f16 convert (grid-stride)
// ---------------------------------------------------------------------------
__global__ void cvt_f32_to_f16(const float* __restrict__ in,
                               _Float16* __restrict__ out, long long n) {
    long long i = (long long)blockIdx.x * blockDim.x + threadIdx.x;
    long long stride = (long long)gridDim.x * blockDim.x;
    for (; i < n; i += stride) out[i] = (_Float16)in[i];
}

// fp32 [K][N] -> f16 [N][K] (weight transpose; contiguous reads)
__global__ void cvt_transpose_f16(const float* __restrict__ in,
                                  _Float16* __restrict__ out, int K, int N) {
    long long total = (long long)K * N;
    long long i = (long long)blockIdx.x * blockDim.x + threadIdx.x;
    long long stride = (long long)gridDim.x * blockDim.x;
    for (; i < total; i += stride) {
        int k = (int)(i / N), n = (int)(i % N);
        out[(size_t)n * K + k] = (_Float16)in[i];
    }
}

// ---------------------------------------------------------------------------
// WMMA GEMM with transposed B: C[M,N] = A[M,K] * Bt[N,K]^T, fp32 accum.
// Wave tile 32x32 (4 WMMAs / K-step, 8 b128 loads / K-step).
// Block = 8 waves (2 x 4) = 64 x 128 tile.
// OUT_MODE: 0 = f32 row-major, 1 = f16 row-major, 2 = f16 V-transposed
//           ([B,S,KH*HD] logical -> [B,KH,HD,S] physical).
// ---------------------------------------------------------------------------
template <int OUT_MODE>
__global__ __launch_bounds__(256)
void gemm_f16_wmma_bt(const _Float16* __restrict__ A, const _Float16* __restrict__ Bt,
                      void* __restrict__ C, int M, int N, int K,
                      int lda, int ldc) {
    const int lane    = threadIdx.x & 31;
    const int wave    = threadIdx.x >> 5;
    const int lq      = lane & 15;
    const int halfSel = lane >> 4;

    const int nw = blockIdx.x * 128 + (wave & 3) * 32;   // wave N base
    const int mw = blockIdx.y * 64  + (wave >> 2) * 32;  // wave M base
    if (nw >= N || mw >= M) return;

    // A-frag: lane<16 holds row=lane, K {0..15}; lane>=16 K {16..31} contiguous
    const int kOffA = halfSel * 16;
    // B-frag: lane holds col=lane&15, contraction K {0..15} / {16..31} contiguous
    const int kOffB = halfSel * 16;

    const _Float16* aBase[2], *bBase[2];
    #pragma unroll
    for (int t = 0; t < 2; ++t) {
        aBase[t] = A  + (size_t)(mw + t * 16 + lq) * lda + kOffA;
        bBase[t] = Bt + (size_t)(nw + t * 16 + lq) * K   + kOffB;
    }

    v8f acc[2][2];
    #pragma unroll
    for (int i = 0; i < 2; ++i)
        #pragma unroll
        for (int j = 0; j < 2; ++j)
            acc[i][j] = (v8f){0.f,0.f,0.f,0.f,0.f,0.f,0.f,0.f};

    for (int k0 = 0; k0 < K; k0 += 32) {
        AFrag a[2], b[2];
        #pragma unroll
        for (int t = 0; t < 2; ++t) {
            a[t].h[0] = *reinterpret_cast<const v8h*>(aBase[t] + k0);       // K +0..7
            a[t].h[1] = *reinterpret_cast<const v8h*>(aBase[t] + k0 + 8);   // K +8..15
            b[t].h[0] = *reinterpret_cast<const v8h*>(bBase[t] + k0);       // K +0..7
            b[t].h[1] = *reinterpret_cast<const v8h*>(bBase[t] + k0 + 8);   // K +8..15
        }
        if (k0 + 32 < K) {
            __builtin_prefetch(aBase[0] + k0 + 32, 0, 0);  // global_prefetch_b8
            __builtin_prefetch(bBase[0] + k0 + 32, 0, 0);
        }
        #pragma unroll
        for (int i = 0; i < 2; ++i)
            #pragma unroll
            for (int j = 0; j < 2; ++j)
                acc[i][j] = __builtin_amdgcn_wmma_f32_16x16x32_f16(
                    false, a[i].v, false, b[j].v, (short)0, acc[i][j], false, false);
    }

    // C layout: VGPR i -> row base+i (lanes 0-15) / base+8+i (lanes 16-31)
    #pragma unroll
    for (int i = 0; i < 2; ++i) {
        const int rowC = mw + i * 16 + halfSel * 8;
        #pragma unroll
        for (int j = 0; j < 2; ++j) {
            const int colC = nw + j * 16 + lq;
            #pragma unroll
            for (int e = 0; e < 8; ++e) {
                const int m = rowC + e;
                if (OUT_MODE == 0) {
                    ((float*)C)[(size_t)m * ldc + colC] = acc[i][j][e];
                } else if (OUT_MODE == 1) {
                    ((_Float16*)C)[(size_t)m * ldc + colC] = (_Float16)acc[i][j][e];
                } else {
                    // V output: m=(b,s), colC=(kh,hd) -> [B][KH][HD][S]
                    const int bb = m >> 11, ss = m & (S_ - 1);     // S_ == 2048
                    const int khh = colC >> 7, hd = colC & (HD_ - 1);
                    ((_Float16*)C)[(((size_t)bb * KH_ + khh) * HD_ + hd) * S_ + ss] =
                        (_Float16)acc[i][j][e];
                }
            }
        }
    }
}

// ---------------------------------------------------------------------------
// RoPE (interleaved pairs, matches reference): in-place on f16 [B,S,nh,HD]
// ---------------------------------------------------------------------------
__global__ void rope_f16(_Float16* __restrict__ x, const int* __restrict__ pos_ids,
                         int nh, long long total_pairs) {
    long long idx = (long long)blockIdx.x * blockDim.x + threadIdx.x;
    long long stride = (long long)gridDim.x * blockDim.x;
    const float logTheta = 9.210340371976184f; // ln(10000)
    for (; idx < total_pairs; idx += stride) {
        int j    = (int)(idx % (HD_ / 2));
        long long r = idx / (HD_ / 2);
        int head = (int)(r % nh);  r /= nh;
        int s    = (int)(r % S_);
        int b    = (int)(r / S_);
        float pos = (float)pos_ids[(size_t)b * S_ + s];
        float inv_freq = __expf(-((2.0f * j) / (float)HD_) * logTheta);
        float ang = pos * inv_freq;
        float c = __cosf(ang), sn = __sinf(ang);
        size_t base = (((size_t)b * S_ + s) * nh + head) * HD_ + 2 * j;
        float x0 = (float)x[base], x1 = (float)x[base + 1];
        x[base]     = (_Float16)(x0 * c - x1 * sn);
        x[base + 1] = (_Float16)(x0 * sn + x1 * c);
    }
}

// ---------------------------------------------------------------------------
// Flash attention: block = (b, head, 64 q rows), 4 waves x 16 q rows.
// Per 32-key step: 8 WMMA for QK^T, online softmax, LDS transpose of P,
// 8 WMMA for P*V (V pre-transposed: [B,KH,HD,S]). GQA: kv head = head/4.
// ---------------------------------------------------------------------------
__global__ __launch_bounds__(128)
void flash_attn_wmma(const _Float16* __restrict__ q, const _Float16* __restrict__ k,
                     const _Float16* __restrict__ vt, const int* __restrict__ amask,
                     _Float16* __restrict__ out) {
    const int lane    = threadIdx.x & 31;
    const int wave    = threadIdx.x >> 5;
    const int lq      = lane & 15;
    const int halfSel = lane >> 4;

    const int b    = blockIdx.z;
    const int head = blockIdx.y;
    const int kh   = head >> 2;                  // H_/KH_ == 4
    const int q0   = blockIdx.x * 64 + wave * 16;

    alignas(16) __shared__ _Float16 pSh[4][16 * 32];  // per-wave P staging

    // ---- load Q fragments (16 rows x HD=128) once; fold 1/sqrt(HD) into Q
    const _Float16 scaleH = (_Float16)0.08838834764831845f;  // 1/sqrt(128)
    const _Float16* qRow = q + (((size_t)b * S_ + (q0 + lq)) * H_ + head) * HD_;
    AFrag aQ[4];
    #pragma unroll
    for (int t = 0; t < 4; ++t) {
        const int base = t * 32 + halfSel * 16;
        aQ[t].h[0] = *reinterpret_cast<const v8h*>(qRow + base);
        aQ[t].h[1] = *reinterpret_cast<const v8h*>(qRow + base + 8);
        #pragma unroll
        for (int e = 0; e < 16; ++e) aQ[t].v[e] = aQ[t].v[e] * scaleH;
    }

    v8f o[8];
    #pragma unroll
    for (int t = 0; t < 8; ++t) o[t] = (v8f){0.f,0.f,0.f,0.f,0.f,0.f,0.f,0.f};
    float mrow[8], lrow[8];
    #pragma unroll
    for (int i = 0; i < 8; ++i) { mrow[i] = -__builtin_inff(); lrow[i] = 0.f; }

    const int kEnd = min(S_, (int)blockIdx.x * 64 + 64);   // causal upper bound

    for (int kt = 0; kt < kEnd; kt += 32) {
        // pad mask for this lane's two key columns (hoisted out of row loop)
        const bool am0 = amask[(size_t)b * S_ + kt + lq] != 0;
        const bool am1 = amask[(size_t)b * S_ + kt + 16 + lq] != 0;

        // ---- scores: two 16x16 tiles over 32 keys
        v8f s[2];
        #pragma unroll
        for (int tile = 0; tile < 2; ++tile) {
            s[tile] = (v8f){0.f,0.f,0.f,0.f,0.f,0.f,0.f,0.f};
            const int key = kt + tile * 16 + lq;     // lane's column = key
            const _Float16* kRow = k + (((size_t)b * S_ + key) * KH_ + kh) * HD_;
            #pragma unroll
            for (int t = 0; t < 4; ++t) {
                AFrag bk;                             // B = K^T (hd x key)
                const int off = t * 32 + halfSel * 16;
                bk.h[0] = *reinterpret_cast<const v8h*>(kRow + off);
                bk.h[1] = *reinterpret_cast<const v8h*>(kRow + off + 8);
                s[tile] = __builtin_amdgcn_wmma_f32_16x16x32_f16(
                    false, aQ[t].v, false, bk.v, (short)0, s[tile], false, false);
            }
        }

        // ---- mask + online softmax (row reductions via wave32 shuffles)
        float alpha[8];
        #pragma unroll
        for (int i = 0; i < 8; ++i) {
            const int row = q0 + halfSel * 8 + i;
            const bool ok0 = am0 && (kt + lq <= row);
            const bool ok1 = am1 && (kt + 16 + lq <= row);
            float v0 = ok0 ? s[0][i] : -1.0e30f;
            float v1 = ok1 ? s[1][i] : -1.0e30f;
            float mnew = fmaxf(mrow[i], rowMax16(fmaxf(v0, v1)));
            alpha[i] = __expf(mrow[i] - mnew);       // 0 on first iteration
            float p0 = ok0 ? __expf(v0 - mnew) : 0.f;
            float p1 = ok1 ? __expf(v1 - mnew) : 0.f;
            lrow[i] = lrow[i] * alpha[i] + rowSum16(p0 + p1);
            mrow[i] = mnew;
            // stage P (row-major 16x32) for the A-layout reload
            pSh[wave][(halfSel * 8 + i) * 32 + lq]      = (_Float16)p0;
            pSh[wave][(halfSel * 8 + i) * 32 + 16 + lq] = (_Float16)p1;
        }
        #pragma unroll
        for (int t = 0; t < 8; ++t)
            #pragma unroll
            for (int i = 0; i < 8; ++i) o[t][i] *= alpha[i];

        // wave-synchronous LDS transpose: drain DS ops before cross-lane read
        asm volatile("s_wait_dscnt 0" ::: "memory");

        AFrag pA;                                     // P as 16x32 A-matrix
        pA.h[0] = *reinterpret_cast<const v8h*>(&pSh[wave][lq * 32 + halfSel * 16]);
        pA.h[1] = *reinterpret_cast<const v8h*>(&pSh[wave][lq * 32 + halfSel * 16 + 8]);

        // ---- O += P(16x32) * V(32xHD): V is [B,KH,HD,S] -> contiguous keys
        #pragma unroll
        for (int t = 0; t < 8; ++t) {
            const int hdcol = t * 16 + lq;
            const _Float16* vRow =
                vt + (((size_t)b * KH_ + kh) * HD_ + hdcol) * S_ + kt + halfSel * 16;
            AFrag bV;
            bV.h[0] = *reinterpret_cast<const v8h*>(vRow);       // keys +0..7
            bV.h[1] = *reinterpret_cast<const v8h*>(vRow + 8);   // keys +8..15
            o[t] = __builtin_amdgcn_wmma_f32_16x16x32_f16(
                false, pA.v, false, bV.v, (short)0, o[t], false, false);
        }
        asm volatile("s_wait_dscnt 0" ::: "memory");   // before next-iter P stores
    }

    // ---- epilogue: normalize and store f16 [B,S,H,HD]
    #pragma unroll
    for (int i = 0; i < 8; ++i) {
        const int row = q0 + halfSel * 8 + i;
        const float inv = 1.0f / lrow[i];
        _Float16* dst = out + (((size_t)b * S_ + row) * H_ + head) * HD_;
        #pragma unroll
        for (int t = 0; t < 8; ++t)
            dst[t * 16 + lq] = (_Float16)(o[t][i] * inv);
    }
}

// ---------------------------------------------------------------------------
// Host-side orchestration
// ---------------------------------------------------------------------------
extern "C" void kernel_launch(void* const* d_in, const int* in_sizes, int n_in,
                              void* d_out, int out_size, void* d_ws, size_t ws_size,
                              hipStream_t stream) {
    const float* hidden = (const float*)d_in[0];
    const int*   amask  = (const int*)d_in[1];
    const int*   posids = (const int*)d_in[2];
    const float* wq     = (const float*)d_in[3];
    const float* wk     = (const float*)d_in[4];
    const float* wv     = (const float*)d_in[5];
    const float* wo     = (const float*)d_in[6];
    float*       outp   = (float*)d_out;

    const long long nHid = (long long)B_ * S_ * D_;          // 8M
    const long long nWq  = (long long)D_ * H_ * HD_;         // 4M
    const long long nWk  = (long long)D_ * KH_ * HD_;        // 1M
    const long long nQ   = (long long)B_ * S_ * H_ * HD_;    // 8M
    const long long nK   = (long long)B_ * S_ * KH_ * HD_;   // 2M

    _Float16* p = (_Float16*)d_ws;
    _Float16* hid_h  = p;  p += nHid;
    _Float16* wqT    = p;  p += nWq;   // [H*HD][D]
    _Float16* wkT    = p;  p += nWk;   // [KH*HD][D]
    _Float16* wvT    = p;  p += nWk;
    _Float16* woT    = p;  p += nWq;   // [D][H*HD]
    _Float16* q_h    = p;  p += nQ;    // [B,S,H,HD]
    _Float16* k_h    = p;  p += nK;    // [B,S,KH,HD]
    _Float16* v_t    = p;  p += nK;    // [B,KH,HD,S]
    _Float16* attn_h = p;  p += nQ;    // [B,S,H*HD]
    (void)ws_size; (void)n_in; (void)in_sizes; (void)out_size;

    const int CT = 256;
    // 1) convert hidden; convert+transpose weights
    cvt_f32_to_f16<<<2048, CT, 0, stream>>>(hidden, hid_h, nHid);
    cvt_transpose_f16<<<2048, CT, 0, stream>>>(wq, wqT, D_, H_ * HD_);
    cvt_transpose_f16<<<1024, CT, 0, stream>>>(wk, wkT, D_, KH_ * HD_);
    cvt_transpose_f16<<<1024, CT, 0, stream>>>(wv, wvT, D_, KH_ * HD_);
    cvt_transpose_f16<<<2048, CT, 0, stream>>>(wo, woT, H_ * HD_, D_);

    // 2) QKV projections (WMMA GEMM, Bt layout)
    const int M = B_ * S_;                   // 4096
    {
        dim3 g((H_ * HD_) / 128, M / 64);    // 16 x 64
        gemm_f16_wmma_bt<1><<<g, 256, 0, stream>>>(hid_h, wqT, q_h,
                                                   M, H_ * HD_, D_, D_, H_ * HD_);
    }
    {
        dim3 g((KH_ * HD_) / 128, M / 64);   // 4 x 64
        gemm_f16_wmma_bt<1><<<g, 256, 0, stream>>>(hid_h, wkT, k_h,
                                                   M, KH_ * HD_, D_, D_, KH_ * HD_);
        gemm_f16_wmma_bt<2><<<g, 256, 0, stream>>>(hid_h, wvT, v_t,
                                                   M, KH_ * HD_, D_, D_, 0);
    }

    // 3) RoPE on Q and K
    rope_f16<<<2048, 256, 0, stream>>>(q_h, posids, H_,  nQ / 2);
    rope_f16<<<1024, 256, 0, stream>>>(k_h, posids, KH_, nK / 2);

    // 4) flash attention
    {
        dim3 g(S_ / 64, H_, B_);             // 32 x 16 x 2
        flash_attn_wmma<<<g, 128, 0, stream>>>(q_h, k_h, v_t, amask, attn_h);
    }

    // 5) output projection (WMMA GEMM, f32 out)
    {
        dim3 g(D_ / 128, M / 64);            // 16 x 64
        gemm_f16_wmma_bt<0><<<g, 256, 0, stream>>>(attn_h, woT, outp,
                                                   M, D_, H_ * HD_, H_ * HD_, D_);
    }
}
